// ClusteringModule_79216376808118
// MI455X (gfx1250) — compile-verified, matrix operand order checked by reading, not run on hardware
//
#include <hip/hip_runtime.h>
#include <hip/hip_bf16.h>
#include <stdint.h>

typedef __attribute__((ext_vector_type(16))) _Float16 v16h;
typedef __attribute__((ext_vector_type(8)))  _Float16 v8h;
typedef __attribute__((ext_vector_type(8)))  float    v8f;
typedef __attribute__((ext_vector_type(4)))  float    v4f;

// ---------------------------------------------------------------------------
// Problem constants (from reference)
// ---------------------------------------------------------------------------
#define B_     16
#define SN_    64
#define C_     480
#define T_     2
#define FY_    60
#define FX_    80
#define OS_    8
#define DIN_   (C_ * OS_ * OS_)   // 30720
#define NDF1_  1024
#define NDF2_  256
#define K_     64
#define ROWS_  (B_ * SN_)         // 1024

// ---------------------------------------------------------------------------
// CDNA5 async global->LDS copy (VFLAT encoding, tracked by ASYNCcnt).
// One instruction moves 32 lanes x 16B = 512B straight into LDS,
// no VGPR round-trip, no loadcnt stall.
// ---------------------------------------------------------------------------
__device__ __forceinline__ void async_ld16(uint32_t lds_byte_off, const void* gaddr)
{
    asm volatile("global_load_async_to_lds_b128 %0, %1, off"
                 :: "v"(lds_byte_off), "v"(gaddr)
                 : "memory");
}
__device__ __forceinline__ void wait_async0()
{
    asm volatile("s_wait_asynccnt 0x0" ::: "memory");
}

// ---------------------------------------------------------------------------
// Kernel 0: f32 -> f16 conversion (one pass; lets both GEMM operands use the
// async-to-LDS path and halves the L2-resident weight footprint).
// ---------------------------------------------------------------------------
__global__ __launch_bounds__(256)
void cvt_f32_f16_kernel(const float* __restrict__ src,
                        _Float16* __restrict__ dst, int n4)
{
    const int i = blockIdx.x * blockDim.x + threadIdx.x;
    if (i < n4) {
        const v4f v = ((const v4f*)src)[i];
        _Float16* d = dst + (size_t)i * 4;
        d[0] = (_Float16)v[0]; d[1] = (_Float16)v[1];
        d[2] = (_Float16)v[2]; d[3] = (_Float16)v[3];
    }
}

// ---------------------------------------------------------------------------
// Kernel 1: ROI-align + pack pooled features as f16 matrix A (1024 x 30720)
// ---------------------------------------------------------------------------
__global__ __launch_bounds__(256)
void roi_pack_kernel(const float* __restrict__ z,
                     const float* __restrict__ boxes,
                     _Float16* __restrict__ A)
{
    const int blk = blockIdx.x;          // 0..1023
    const int b   = blk >> 6;
    const int s   = blk & 63;
    const int tid = threadIdx.x;

    __shared__ float w0s[64], w1s[64], w2s[64], w3s[64];
    __shared__ int   o0s[64], o1s[64], o2s[64], o3s[64];

    if (tid < 64) {
        const float* bb = &boxes[((size_t)b * SN_ + s) * 4];
        float x1 = bb[0] * 0.125f, y1 = bb[1] * 0.125f;
        float x2 = bb[2] * 0.125f, y2 = bb[3] * 0.125f;
        x1 = (x1 != x1) ? 0.0f : x1;  y1 = (y1 != y1) ? 0.0f : y1;
        x2 = (x2 != x2) ? 0.0f : x2;  y2 = (y2 != y2) ? 0.0f : y2;
        x1 -= 0.5f; y1 -= 0.5f; x2 -= 0.5f; y2 -= 0.5f;   // ALIGNED offset
        const float bw = (x2 - x1) * (1.0f / OS_);
        const float bh = (y2 - y1) * (1.0f / OS_);
        const int ox = tid & 7, oy = tid >> 3;
        const float X = x1 + ((float)ox + 0.5f) * bw;
        const float Y = y1 + ((float)oy + 0.5f) * bh;
        const bool valid = (Y >= -1.0f) && (Y <= (float)FY_) &&
                           (X >= -1.0f) && (X <= (float)FX_);
        float y = fminf(fmaxf(Y, 0.0f), (float)(FY_ - 1));
        float x = fminf(fmaxf(X, 0.0f), (float)(FX_ - 1));
        const float y0f = floorf(y), x0f = floorf(x);
        const int y0 = (int)y0f, x0 = (int)x0f;
        const int y1i = min(y0 + 1, FY_ - 1);
        const int x1i = min(x0 + 1, FX_ - 1);
        const float ly = y - y0f, lx = x - x0f;
        const float hy = 1.0f - ly, hx = 1.0f - lx;
        const float m = valid ? 1.0f : 0.0f;
        w0s[tid] = hy * hx * m;  w1s[tid] = hy * lx * m;
        w2s[tid] = ly * hx * m;  w3s[tid] = ly * lx * m;
        o0s[tid] = y0 * FX_ + x0;   o1s[tid] = y0 * FX_ + x1i;
        o2s[tid] = y1i * FX_ + x0;  o3s[tid] = y1i * FX_ + x1i;
    }
    __syncthreads();

    const float* feat = z + (size_t)b * (C_ * T_ * FY_ * FX_) + (size_t)(T_ - 1) * (FY_ * FX_);
    _Float16* arow = A + (size_t)blk * DIN_;

    #pragma unroll 4
    for (int it = 0; it < DIN_ / 256; ++it) {
        const int idx = it * 256 + tid;
        const int c = idx >> 6;
        const int p = idx & 63;
        const float* f = feat + (size_t)c * (T_ * FY_ * FX_);
        const float v = w0s[p] * f[o0s[p]] + w1s[p] * f[o1s[p]] +
                        w2s[p] * f[o2s[p]] + w3s[p] * f[o3s[p]];
        arow[idx] = (_Float16)v;
    }
}

// ---------------------------------------------------------------------------
// Kernel 2/3: NT GEMM  C[M,N] = A[M,K] * W[N,K]^T + bias, both f16 row-major.
//
// Block: 128 threads = 4 waves (2x2). Block tile 64(M) x 128(N).
// Wave tile 32x64 = 2x4 WMMA tiles -> 16 v_wmma_f32_16x16x32_f16 per K-step.
// K staged in steps of 64 through double-buffered LDS via
// GLOBAL_LOAD_ASYNC_TO_LDS_B128 (ASYNCcnt), pitch 80 halves = 160B so every
// fragment is 16B-aligned => ds_load_b128.
//
// Both operands are K-contiguous row-major, so the A-matrix per-lane layout
// (lane l: row = l&15, K-half = (l>>4)*8, halves {0..7, 16..23}) serves the
// B operand as well (roles M<->N swapped).
// ---------------------------------------------------------------------------
#define BTM   64      // block tile M
#define BTN   128     // block tile N
#define GKS   64      // K step
#define LPITCH 80     // LDS pitch in halves (160B)
#define ABUF_B (BTM * LPITCH * 2)   // bytes per A buffer
#define WBUF_B (BTN * LPITCH * 2)   // bytes per W buffer

__global__ __launch_bounds__(128)
void gemm_nt_wmma_kernel(const _Float16* __restrict__ A,
                         const _Float16* __restrict__ W,
                         const float* __restrict__ bias,
                         _Float16* __restrict__ outH,
                         float* __restrict__ outF,
                         int M, int N, int K)
{
    __shared__ _Float16 lA[2][BTM * LPITCH];
    __shared__ _Float16 lW[2][BTN * LPITCH];

    const int tid  = threadIdx.x;
    const int lane = tid & 31;
    const int wid  = tid >> 5;
    const int wm   = wid & 1;         // wave row: 32 M-rows each
    const int wn   = wid >> 1;        // wave col: 64 N-cols each
    const int bm   = blockIdx.y * BTM;
    const int bn   = blockIdx.x * BTN;

    const int mfrag = lane & 15;        // row/col within a 16x16 tile
    const int khalf = (lane >> 4) * 8;  // K sub-offset: 0 or 8

    // ---- precompute per-lane async-copy addresses (16B chunks) -------------
    // A tile: 64 rows x 64 halves = 512 chunks -> 4 per lane
    uint32_t ldsAoff[4]; const _Float16* gAp[4];
    #pragma unroll
    for (int i = 0; i < 4; ++i) {
        const int chunk = (i * 4 + wid) * 32 + lane;
        const int r = chunk >> 3;
        const int c = (chunk & 7) * 8;
        ldsAoff[i] = (uint32_t)(r * LPITCH + c) * 2u;
        gAp[i] = A + (size_t)(bm + r) * K + c;
    }
    // W tile: 128 rows x 64 halves = 1024 chunks -> 8 per lane
    uint32_t ldsWoff[8]; const _Float16* gWp[8];
    #pragma unroll
    for (int i = 0; i < 8; ++i) {
        const int chunk = (i * 4 + wid) * 32 + lane;
        const int r = chunk >> 3;
        const int c = (chunk & 7) * 8;
        ldsWoff[i] = (uint32_t)(r * LPITCH + c) * 2u;
        gWp[i] = W + (size_t)(bn + r) * K + c;
    }
    const uint32_t baseA = (uint32_t)(uintptr_t)&lA[0][0];
    const uint32_t baseW = (uint32_t)(uintptr_t)&lW[0][0];

    v8f acc[2][4] = {};

    // ---- prologue: async-stage tile 0 into buffer 0 ------------------------
    #pragma unroll
    for (int i = 0; i < 4; ++i) async_ld16(baseA + ldsAoff[i], gAp[i]);
    #pragma unroll
    for (int i = 0; i < 8; ++i) async_ld16(baseW + ldsWoff[i], gWp[i]);
    wait_async0();
    __syncthreads();

    int parity = 0;
    for (int k0 = 0; k0 < K; k0 += GKS) {
        // ---- issue async copies for the NEXT tile into the other buffer ----
        const int nk = k0 + GKS;
        if (nk < K) {
            const uint32_t bA = baseA + (parity ^ 1) * ABUF_B;
            const uint32_t bW = baseW + (parity ^ 1) * WBUF_B;
            #pragma unroll
            for (int i = 0; i < 4; ++i) async_ld16(bA + ldsAoff[i], gAp[i] + nk);
            #pragma unroll
            for (int i = 0; i < 8; ++i) async_ld16(bW + ldsWoff[i], gWp[i] + nk);
        }

        // ---- compute on the CURRENT buffer ---------------------------------
        const _Float16* sA = lA[parity];
        const _Float16* sW = lW[parity];
        #pragma unroll
        for (int ks = 0; ks < GKS; ks += 32) {
            v16h af[2], bf[4];
            #pragma unroll
            for (int i = 0; i < 2; ++i) {
                const _Float16* p = &sA[(wm * 32 + i * 16 + mfrag) * LPITCH + ks + khalf];
                *(v8h*)&af[i]       = *(const v8h*)p;        // K {0..7 | 8..15}
                *((v8h*)&af[i] + 1) = *(const v8h*)(p + 16); // K {16..23 | 24..31}
            }
            #pragma unroll
            for (int j = 0; j < 4; ++j) {
                const _Float16* p = &sW[(wn * 64 + j * 16 + mfrag) * LPITCH + ks + khalf];
                *(v8h*)&bf[j]       = *(const v8h*)p;
                *((v8h*)&bf[j] + 1) = *(const v8h*)(p + 16);
            }
            #pragma unroll
            for (int i = 0; i < 2; ++i)
                #pragma unroll
                for (int j = 0; j < 4; ++j)
                    acc[i][j] = __builtin_amdgcn_wmma_f32_16x16x32_f16(
                        false, af[i], false, bf[j],
                        (short)0, acc[i][j], false, false);
        }

        // next buffer fully landed + all waves done reading current buffer
        wait_async0();
        __syncthreads();
        parity ^= 1;
    }

    // ---- writeback (+bias). D layout: VGPR r of lane l holds
    // row = i*16 + (l>>4)*8 + r, col = j*16 + (l&15).
    #pragma unroll
    for (int i = 0; i < 2; ++i) {
        #pragma unroll
        for (int j = 0; j < 4; ++j) {
            const int col = bn + wn * 64 + j * 16 + (lane & 15);
            const float bc = bias ? bias[col] : 0.0f;
            #pragma unroll
            for (int r = 0; r < 8; ++r) {
                const int row = bm + wm * 32 + i * 16 + ((lane >> 4) * 8) + r;
                const float v = acc[i][j][r] + bc;
                if (outH) outH[(size_t)row * N + col] = (_Float16)v;
                if (outF) outF[(size_t)row * N + col] = v;
            }
        }
    }
}

// ---------------------------------------------------------------------------
// Kernel 4: t-distribution soft assignment + argmax (ALPHA=1 -> 1/(1+||z-c||))
// ---------------------------------------------------------------------------
__global__ __launch_bounds__(64)
void cluster_assign_kernel(const float* __restrict__ zz,
                           const float* __restrict__ cen,
                           float* __restrict__ s_out,
                           float* __restrict__ c_out)
{
    const int row = blockIdx.x;
    const int k   = threadIdx.x;
    const float* zr = zz  + (size_t)row * NDF2_;
    const float* ck = cen + (size_t)k   * NDF2_;

    float zc = 0.0f, z2 = 0.0f, c2 = 0.0f;
    #pragma unroll 8
    for (int d = 0; d < NDF2_; ++d) {
        const float zv = zr[d], cv = ck[d];
        zc += zv * cv; z2 += zv * zv; c2 += cv * cv;
    }
    const float nrm = sqrtf(fmaxf(z2 + c2 - 2.0f * zc, 0.0f));
    const float sk  = 1.0f / (1.0f + nrm);

    __shared__ float red[64];
    __shared__ float ssum;
    __shared__ int   sarg;
    red[k] = sk;
    __syncthreads();
    if (k == 0) {
        float sum = 0.0f, best = -1.0f; int bi = 0;
        #pragma unroll
        for (int i = 0; i < 64; ++i) {
            sum += red[i];
            if (red[i] > best) { best = red[i]; bi = i; }
        }
        ssum = sum; sarg = bi;
    }
    __syncthreads();
    s_out[(size_t)row * K_ + k] = sk / ssum;
    if (k == 0) c_out[row] = (float)sarg;
}

// ---------------------------------------------------------------------------
// Launch
// ---------------------------------------------------------------------------
extern "C" void kernel_launch(void* const* d_in, const int* in_sizes, int n_in,
                              void* d_out, int out_size, void* d_ws, size_t ws_size,
                              hipStream_t stream)
{
    const float* z     = (const float*)d_in[0];  // (16,480,2,60,80)
    const float* boxes = (const float*)d_in[1];  // (16,64,4)
    const float* w1    = (const float*)d_in[2];  // (1024,30720)
    const float* b1    = (const float*)d_in[3];  // (1024,)
    const float* w2    = (const float*)d_in[4];  // (256,1024)
    const float* b2    = (const float*)d_in[5];  // (256,)
    const float* cen   = (const float*)d_in[6];  // (64,256)

    // workspace layout (all 16B aligned)
    char* ws = (char*)d_ws;
    _Float16* A   = (_Float16*)ws;                                   // 60 MiB
    _Float16* W1h = (_Float16*)(ws + (size_t)ROWS_ * DIN_ * 2);      // 60 MiB
    _Float16* e1  = (_Float16*)(ws + (size_t)ROWS_ * DIN_ * 2
                                   + (size_t)NDF1_ * DIN_ * 2);      // 2 MiB
    _Float16* W2h = (_Float16*)((char*)e1 + (size_t)ROWS_ * NDF1_ * 2); // 0.5 MiB

    // output layout: zz (1024*256) | s (1024*64) | c (1024)
    float* zz_out = (float*)d_out;
    float* s_out  = zz_out + (size_t)ROWS_ * NDF2_;
    float* c_out  = s_out  + (size_t)ROWS_ * K_;

    // 0) one-pass weight conversion to f16 (enables async-to-LDS staging)
    {
        const int n4_w1 = (NDF1_ * DIN_) / 4;    // 7,864,320
        const int n4_w2 = (NDF2_ * NDF1_) / 4;   // 65,536
        cvt_f32_f16_kernel<<<(n4_w1 + 255) / 256, 256, 0, stream>>>(w1, W1h, n4_w1);
        cvt_f32_f16_kernel<<<(n4_w2 + 255) / 256, 256, 0, stream>>>(w2, W2h, n4_w2);
    }

    // 1) ROI-align + pack f16 A matrix
    roi_pack_kernel<<<ROWS_, 256, 0, stream>>>(z, boxes, A);

    // 2) e1 = A @ w1^T + b1   (1024 x 30720 x 1024) -> f16
    gemm_nt_wmma_kernel<<<dim3(NDF1_ / BTN, ROWS_ / BTM), 128, 0, stream>>>(
        A, W1h, b1, e1, nullptr, ROWS_, NDF1_, DIN_);

    // 3) zz = e1 @ w2^T + b2  (1024 x 1024 x 256) -> f32 (output)
    gemm_nt_wmma_kernel<<<dim3(NDF2_ / BTN, ROWS_ / BTM), 128, 0, stream>>>(
        e1, W2h, b2, nullptr, zz_out, ROWS_, NDF2_, NDF1_);

    // 4) soft assignment + argmax
    cluster_assign_kernel<<<ROWS_, 64, 0, stream>>>(zz_out, cen, s_out, c_out);
}
